// PhiMoEAttention_2886218023362
// MI455X (gfx1250) — matrix operand order repaired and checked by reading, not compiled, not run on hardware
//
#include <hip/hip_runtime.h>

#define DEV __device__ __forceinline__

typedef __bf16 bf16;
typedef __attribute__((ext_vector_type(16))) bf16 bf16x16;
typedef __attribute__((ext_vector_type(8)))  bf16 bf16x8;
typedef __attribute__((ext_vector_type(8)))  float v8f;

constexpr int BB   = 2;
constexpr int SS   = 1024;
constexpr int HID  = 4096;
constexpr int NH   = 32;
constexpr int NKV  = 8;
constexpr int DH   = 128;
constexpr int GG   = NH / NKV;            // 4
constexpr int WIN  = 512;
constexpr int QKVC = (NH + 2 * NKV) * DH; // 6144
constexpr int KOFF = NH * DH;             // 4096
constexpr int VOFF = (NH + NKV) * DH;     // 5120
constexpr int AT   = NH * DH;             // 4096

// f32 -> bf16 round-to-nearest-even (bit trick; used only outside hot loops)
DEV bf16 tobf(float f) {
  unsigned u = __builtin_bit_cast(unsigned, f);
  unsigned short h = (unsigned short)((u + 0x7fffu + ((u >> 16) & 1u)) >> 16);
  return __builtin_bit_cast(bf16, h);
}
DEV float tof(bf16 h) {
  unsigned short u = __builtin_bit_cast(unsigned short, h);
  unsigned v = ((unsigned)u) << 16;
  return __builtin_bit_cast(float, v);
}

DEV v8f wmma_bf16(bf16x16 a, bf16x16 b, v8f c) {
  return __builtin_amdgcn_wmma_f32_16x16x32_bf16(false, a, false, b, (short)0, c, false, false);
}

// fragment from two 16-byte runs of bf16 (global or LDS)
DEV bf16x16 frag2(const bf16* p0, const bf16* p1) {
  bf16x8 lo = *(const bf16x8*)p0;
  bf16x8 hi = *(const bf16x8*)p1;
  bf16x16 f;
#pragma unroll
  for (int i = 0; i < 8; i++) { f[i] = lo[i]; f[i + 8] = hi[i]; }
  return f;
}

// async copy 32 B (16 bf16) global -> LDS, tracked by ASYNCcnt
DEV void async_cp32(const bf16* g, bf16* l) {
  unsigned loff = (unsigned)(size_t)l;   // flat addr low 32 bits == LDS offset
  asm volatile("global_load_async_to_lds_b128 %0, %1, off\n\t"
               "global_load_async_to_lds_b128 %0, %1, off offset:16"
               :: "v"(loff), "v"(g) : "memory");
}
DEV void wait_async0() {
  asm volatile("s_wait_asynccnt 0x0" ::: "memory");
}

// ---------------------------------------------------------------------------
// elementwise f32 -> bf16 (memory bound, one pass)
// ---------------------------------------------------------------------------
__global__ __launch_bounds__(256) void cvt_f32_bf16(
    const float* __restrict__ src, bf16* __restrict__ dst, int n8)
{
  int i = blockIdx.x * blockDim.x + threadIdx.x;
  if (i >= n8) return;
  const float4* p = (const float4*)(src + (size_t)i * 8);
  float4 a = p[0], b = p[1];
  bf16x8 o;
  o[0]=tobf(a.x); o[1]=tobf(a.y); o[2]=tobf(a.z); o[3]=tobf(a.w);
  o[4]=tobf(b.x); o[5]=tobf(b.y); o[6]=tobf(b.z); o[7]=tobf(b.w);
  *(bf16x8*)(dst + (size_t)i * 8) = o;
}

// ---------------------------------------------------------------------------
// GEMM: C[M,N] = A[M,K] * B[N,K]^T + bias ; A,B bf16 row-major, K contiguous.
// 128x128x32 tile, 8 waves (2x4), 64x32 of C per wave via 16x16x32 WMMA.
// LDS tiles double-buffered, filled with global_load_async_to_lds_b128.
// ---------------------------------------------------------------------------
constexpr int BM = 128, BN = 128, BK = 32, LP = 40; // 80 B padded LDS rows

template <bool OUT_BF16>
__global__ __launch_bounds__(256) void gemm_bias_kernel(
    const bf16* __restrict__ A, const bf16* __restrict__ Bw,
    const float* __restrict__ bias, float* __restrict__ Cf,
    bf16* __restrict__ Cb, int M, int N, int K)
{
  __shared__ __align__(16) bf16 As[2][BM][LP];
  __shared__ __align__(16) bf16 Bs[2][BN][LP];

  const int tid  = threadIdx.x;
  const int lane = tid & 31;
  const int wv   = tid >> 5;
  const int wm   = wv >> 2;        // 0..1
  const int wn   = wv & 3;         // 0..3
  const int m0   = blockIdx.y * BM;
  const int n0   = blockIdx.x * BN;
  const int l15  = lane & 15;
  const int hg   = lane >> 4;
  const int lr   = tid >> 1;       // staged tile row (0..127)
  const int lc   = (tid & 1) * 16; // staged 16-elem column segment

  const bf16* arow = A  + (size_t)(m0 + lr) * K + lc;
  const bf16* brow = Bw + (size_t)(n0 + lr) * K + lc;

  v8f acc[4][2];
#pragma unroll
  for (int ms = 0; ms < 4; ms++)
#pragma unroll
    for (int ns = 0; ns < 2; ns++)
#pragma unroll
      for (int i = 0; i < 8; i++) acc[ms][ns][i] = 0.f;

  const int nk = K / BK;
  // prologue: stage tile 0 into buffer 0
  async_cp32(arow, &As[0][lr][lc]);
  async_cp32(brow, &Bs[0][lr][lc]);

  for (int i = 0; i < nk; i++) {
    const int cur = i & 1;
    wait_async0();        // this wave's current-tile copies done
    __syncthreads();      // whole block's current tile visible; prev reads done

    if (i + 1 < nk) {     // stage next tile under this tile's compute
      async_cp32(arow + (size_t)(i + 1) * BK, &As[1 - cur][lr][lc]);
      async_cp32(brow + (size_t)(i + 1) * BK, &Bs[1 - cur][lr][lc]);
    }

    bf16x16 af[4], bfr[2];
#pragma unroll
    for (int ms = 0; ms < 4; ms++) {
      const bf16* r = &As[cur][wm * 64 + ms * 16 + l15][0];
      af[ms] = frag2(r + hg * 8, r + 16 + hg * 8);        // A-frag K split
    }
#pragma unroll
    for (int ns = 0; ns < 2; ns++) {
      const bf16* r = &Bs[cur][wn * 32 + ns * 16 + l15][0];
      bfr[ns] = frag2(r + hg * 8, r + 16 + hg * 8);       // B-frag K split
    }
#pragma unroll
    for (int ms = 0; ms < 4; ms++)
#pragma unroll
      for (int ns = 0; ns < 2; ns++)
        acc[ms][ns] = wmma_bf16(af[ms], bfr[ns], acc[ms][ns]);
  }

#pragma unroll
  for (int ms = 0; ms < 4; ms++)
#pragma unroll
    for (int ns = 0; ns < 2; ns++) {
      int row = m0 + wm * 64 + ms * 16 + hg * 8;
      int col = n0 + wn * 32 + ns * 16 + l15;
      float bv = bias ? bias[col] : 0.f;
#pragma unroll
      for (int i = 0; i < 8; i++) {
        float v = acc[ms][ns][i] + bv;
        if (OUT_BF16) Cb[(size_t)(row + i) * N + col] = tobf(v);
        else          Cf[(size_t)(row + i) * N + col] = v;
      }
    }
}

// ---------------------------------------------------------------------------
// RoPE in place on q (heads 0..NH-1) and k (heads NH..NH+NKV-1), bf16 storage
// ---------------------------------------------------------------------------
__global__ __launch_bounds__(64) void rope_kernel(
    bf16* __restrict__ qkv, const float* __restrict__ cs, const float* __restrict__ sn)
{
  int idx  = blockIdx.x;
  int head = idx % (NH + NKV);
  int bs   = idx / (NH + NKV);
  int s    = bs % SS;
  int d    = threadIdx.x;              // 0..63
  bf16* p  = qkv + (size_t)bs * QKVC + head * DH;
  float c  = cs[s * DH + d];
  float si = sn[s * DH + d];           // cos/sin[d] == cos/sin[d+64]
  float x1 = tof(p[d]), x2 = tof(p[d + 64]);
  p[d]      = tobf(x1 * c - x2 * si);
  p[d + 64] = tobf(x2 * c + x1 * si);
}

// ---------------------------------------------------------------------------
// Sliding-window causal flash attention, 1 wave per 16-query tile per (b,h).
// All Q/K/V traffic is raw bf16 vectors; only softmax runs on VALU.
// ---------------------------------------------------------------------------
__global__ __launch_bounds__(32) void attn_kernel(
    const bf16* __restrict__ qkv, bf16* __restrict__ ab)
{
  const int q0  = blockIdx.x * 16;
  const int h   = blockIdx.y;
  const int b   = blockIdx.z;
  const int kvh = h / GG;
  const int lane = threadIdx.x;
  const int l15 = lane & 15;
  const int hg  = lane >> 4;

  __shared__ __align__(16) bf16 Ps[16][32];

  const float scl = 0.088388347648318447f; // 1/sqrt(128)

  // resident Q fragments (A-frag: lane = query row q0+l15)
  const bf16* qrow = qkv + (size_t)(b * SS + q0 + l15) * QKVC + h * DH;
  bf16x16 qf[4];
#pragma unroll
  for (int j = 0; j < 4; j++)
    qf[j] = frag2(qrow + j * 32 + hg * 8, qrow + j * 32 + 16 + hg * 8);

  v8f o[8];
#pragma unroll
  for (int j = 0; j < 8; j++)
#pragma unroll
    for (int i = 0; i < 8; i++) o[j][i] = 0.f;
  float mi[8], li[8];
#pragma unroll
  for (int i = 0; i < 8; i++) { mi[i] = -INFINITY; li[i] = 0.f; }

  int lo = q0 - (WIN - 1); if (lo < 0) lo = 0;
  for (int k0 = lo & ~31; k0 <= q0 + 15; k0 += 32) {
    // ---- scores: S = Q K^T (two 16-key tiles) ----
    v8f s0, s1;
#pragma unroll
    for (int i = 0; i < 8; i++) { s0[i] = 0.f; s1[i] = 0.f; }

    int key0 = k0 + l15;      if (key0 > SS - 1) key0 = SS - 1;
    int key1 = k0 + 16 + l15; if (key1 > SS - 1) key1 = SS - 1;
    const bf16* kr0 = qkv + (size_t)(b * SS + key0) * QKVC + KOFF + kvh * DH;
    const bf16* kr1 = qkv + (size_t)(b * SS + key1) * QKVC + KOFF + kvh * DH;
#pragma unroll
    for (int j = 0; j < 4; j++) {
      bf16x16 kf0 = frag2(kr0 + j * 32 + hg * 8, kr0 + j * 32 + 16 + hg * 8);
      bf16x16 kf1 = frag2(kr1 + j * 32 + hg * 8, kr1 + j * 32 + 16 + hg * 8);
      s0 = wmma_bf16(qf[j], kf0, s0);
      s1 = wmma_bf16(qf[j], kf1, s1);
    }

    // ---- online softmax in C layout (row = q0 + hg*8 + i, col = key) ----
#pragma unroll
    for (int i = 0; i < 8; i++) {
      int q  = q0 + hg * 8 + i;
      int ka = k0 + l15;
      int kb = k0 + 16 + l15;
      float sa = ((ka <= q) && (q - ka < WIN)) ? s0[i] * scl : -INFINITY;
      float sb = ((kb <= q) && (q - kb < WIN)) ? s1[i] * scl : -INFINITY;
      float v = fmaxf(sa, sb);
      v = fmaxf(v, __shfl_xor(v, 1, 16));
      v = fmaxf(v, __shfl_xor(v, 2, 16));
      v = fmaxf(v, __shfl_xor(v, 4, 16));
      v = fmaxf(v, __shfl_xor(v, 8, 16));
      float mn  = fmaxf(mi[i], v);
      float rsc = (mi[i] == -INFINITY) ? 0.f : __expf(mi[i] - mn);
      float pa  = (mn == -INFINITY) ? 0.f : __expf(sa - mn);
      float pb  = (mn == -INFINITY) ? 0.f : __expf(sb - mn);
      float rs = pa + pb;
      rs += __shfl_xor(rs, 1, 16);
      rs += __shfl_xor(rs, 2, 16);
      rs += __shfl_xor(rs, 4, 16);
      rs += __shfl_xor(rs, 8, 16);
      li[i] = li[i] * rsc + rs;
      mi[i] = mn;
#pragma unroll
      for (int j = 0; j < 8; j++) o[j][i] *= rsc;
      Ps[hg * 8 + i][l15]      = tobf(pa);
      Ps[hg * 8 + i][16 + l15] = tobf(pb);
    }
    __syncthreads();
    // transpose P (C layout -> A-frag layout) through LDS
    bf16x16 pf = frag2(&Ps[l15][hg * 8], &Ps[l15][16 + hg * 8]);
    __syncthreads();

    // ---- O += P V  (B-frag: lane = d column, K = 32 keys) ----
#pragma unroll
    for (int j = 0; j < 8; j++) {
      bf16x16 vf;
#pragma unroll
      for (int t = 0; t < 16; t++) {
        int key = k0 + (t < 8 ? hg * 8 + t : 16 + hg * 8 + (t - 8)); if (key > SS - 1) key = SS - 1;
        vf[t] = qkv[(size_t)(b * SS + key) * QKVC + VOFF + kvh * DH + j * 16 + l15];
      }
      o[j] = wmma_bf16(pf, vf, o[j]);
    }
  }

  // ---- epilogue: divide by l, write bf16 attn buffer [B,S,NH*D] ----
#pragma unroll
  for (int i = 0; i < 8; i++) {
    float inv = (li[i] > 0.f) ? (1.f / li[i]) : 0.f;
    int q = q0 + hg * 8 + i;
#pragma unroll
    for (int j = 0; j < 8; j++)
      ab[(size_t)(b * SS + q) * AT + h * DH + j * 16 + l15] = tobf(o[j][i] * inv);
  }
}

// ---------------------------------------------------------------------------
extern "C" void kernel_launch(void* const* d_in, const int* in_sizes, int n_in,
                              void* d_out, int out_size, void* d_ws, size_t ws_size,
                              hipStream_t stream)
{
  const float* hidden = (const float*)d_in[0];
  const float* cs     = (const float*)d_in[1];
  const float* sn     = (const float*)d_in[2];
  const float* wqkv   = (const float*)d_in[3];
  const float* bqkv   = (const float*)d_in[4];
  const float* wo     = (const float*)d_in[5];
  const float* bo     = (const float*)d_in[6];
  float* out = (float*)d_out;

  // workspace layout (bf16), peak ~92 MB:
  //   [hidden_bf 16.8MB][qkv_bf 25.2MB][wqkv_bf 50.3MB]
  //   after GEMM1, wqkv_bf region is reused as [ab 16.8MB][wo_bf 33.6MB]
  bf16* hidden_bf = (bf16*)d_ws;
  bf16* qkv_bf    = hidden_bf + (size_t)BB * SS * HID;
  bf16* wqkv_bf   = qkv_bf    + (size_t)BB * SS * QKVC;
  bf16* ab        = wqkv_bf;                       // alias (post-GEMM1)
  bf16* wo_bf     = ab + (size_t)BB * SS * AT;     // alias (post-GEMM1)

  const int nHid  = BB * SS * HID / 8;     // 1048576
  const int nWqkv = QKVC * HID / 8;        // 3145728
  const int nWo   = HID * AT / 8;          // 2097152

  // 0) one-pass f32 -> bf16 conversions needed for GEMM1
  cvt_f32_bf16<<<dim3(nHid  / 256), dim3(256), 0, stream>>>(hidden, hidden_bf, nHid);
  cvt_f32_bf16<<<dim3(nWqkv / 256), dim3(256), 0, stream>>>(wqkv,   wqkv_bf,  nWqkv);

  // 1) QKV projection + bias -> bf16
  gemm_bias_kernel<true><<<dim3(QKVC / BN, (BB * SS) / BM), dim3(256), 0, stream>>>(
      hidden_bf, wqkv_bf, bqkv, nullptr, qkv_bf, BB * SS, QKVC, HID);

  // 2) convert w_o (into region that aliased wqkv_bf; safe after GEMM1)
  cvt_f32_bf16<<<dim3(nWo / 256), dim3(256), 0, stream>>>(wo, wo_bf, nWo);

  // 3) RoPE on q and k heads in place
  rope_kernel<<<dim3(BB * SS * (NH + NKV)), dim3(64), 0, stream>>>(qkv_bf, cs, sn);

  // 4) sliding-window causal attention -> bf16
  attn_kernel<<<dim3(SS / 16, NH, BB), dim3(32), 0, stream>>>(qkv_bf, ab);

  // 5) output projection + bias -> f32 result
  gemm_bias_kernel<false><<<dim3(HID / BN, (BB * SS) / BM), dim3(256), 0, stream>>>(
      ab, wo_bf, bo, out, nullptr, BB * SS, HID, AT);
}
